// SAtt_sq_80324478370351
// MI455X (gfx1250) — matrix operand order repaired and checked by reading, not compiled, not run on hardware
//
#include <hip/hip_runtime.h>
#include <hip/hip_bf16.h>

// ---------------------------------------------------------------------------
// Problem constants (reference: B,C,H,W = 4,256,64,64)
// ---------------------------------------------------------------------------
#define B_  4
#define C_  256
#define H_  64
#define W_  64
#define N_  (H_ * W_)      // 4096
#define K9_ (C_ * 9)       // 2304 : implicit-GEMM K for 3x3 conv

typedef __bf16 bf16;
typedef __attribute__((ext_vector_type(16))) __bf16 v16bf;
typedef __attribute__((ext_vector_type(8)))  __bf16 v8bf;
typedef __attribute__((ext_vector_type(8)))  float  v8f;
typedef __attribute__((ext_vector_type(4)))  unsigned v4u;
typedef __attribute__((ext_vector_type(8)))  int      v8i;
typedef __attribute__((ext_vector_type(4)))  int      v4i;

#if __has_builtin(__builtin_amdgcn_tensor_load_to_lds)
#define HAVE_TDM 1
#else
#define HAVE_TDM 0
#endif

__device__ __forceinline__ v8f wmma_bf16(v16bf a, v16bf b, v8f c) {
  // v_wmma_f32_16x16x32_bf16  (neg_a, A, neg_b, B, c_mod, C, reuse_a, reuse_b)
  return __builtin_amdgcn_wmma_f32_16x16x32_bf16(false, a, false, b, (short)0, c,
                                                 false, false);
}

// Async global->LDS copies (gfx1250 ASYNCcnt path). ISA 08_async_tensor §4.4:
// INST_OFFSET is added to BOTH the LDS and global address.
__device__ __forceinline__ void async_copy32(void* lds_ptr, const void* gptr) {
  unsigned lds = (unsigned)(unsigned long long)lds_ptr;   // addrspace(3) offset
  unsigned long long ga = (unsigned long long)gptr;
  asm volatile(
      "global_load_async_to_lds_b128 %0, %1, off\n\t"
      "global_load_async_to_lds_b128 %0, %1, off offset:16"
      :: "v"(lds), "v"(ga) : "memory");
}

__device__ __forceinline__ void async_copy64(void* lds_ptr, const void* gptr) {
  unsigned lds = (unsigned)(unsigned long long)lds_ptr;
  unsigned long long ga = (unsigned long long)gptr;
  asm volatile(
      "global_load_async_to_lds_b128 %0, %1, off\n\t"
      "global_load_async_to_lds_b128 %0, %1, off offset:16\n\t"
      "global_load_async_to_lds_b128 %0, %1, off offset:32\n\t"
      "global_load_async_to_lds_b128 %0, %1, off offset:48"
      :: "v"(lds), "v"(ga) : "memory");
}

__device__ __forceinline__ void wait_async0() {
  asm volatile("s_wait_asynccnt 0x0" ::: "memory");
}

// ---------------------------------------------------------------------------
// Tensor Data Mover: 2D bf16 tile load, descriptor per ISA 08 §8.3/8.4.
// LDS destination rows padded: pad_amount dwords after every pad_interval
// dwords (codes per D# group1).  One wave issues the whole tile.
// ---------------------------------------------------------------------------
__device__ __forceinline__ void tdm_load_2d(void* lds_ptr, const void* gptr,
                                            unsigned tile_d0, unsigned tile_d1,
                                            unsigned tensor_d0, unsigned tensor_d1,
                                            unsigned stride_elems,
                                            unsigned pad_interval_code,
                                            unsigned pad_amount_code) {
  unsigned long long ga = (unsigned long long)gptr;
  v4u g0 = {};
  g0[0] = 1u;                                         // count=1 (user D#)
  g0[1] = (unsigned)(unsigned long long)lds_ptr;      // lds_addr (bytes)
  g0[2] = (unsigned)(ga & 0xffffffffu);               // global_addr[31:0]
  g0[3] = (unsigned)((ga >> 32) & 0x01ffffffu) | (2u << 30);  // [56:32] | type=2
  v8i g1 = {};
  g1[0] = (int)((1u << 16)                            // data_size: 2 bytes
              | (1u << 20)                            // pad_enable
              | (pad_interval_code << 22)
              | (pad_amount_code << 25));
  g1[1] = (int)((tensor_d0 & 0xffffu) << 16);                       // dim0 lo16
  g1[2] = (int)(((tensor_d0 >> 16) & 0xffffu)
              | ((tensor_d1 & 0xffffu) << 16));                     // dim0 hi|dim1 lo
  g1[3] = (int)(((tensor_d1 >> 16) & 0xffffu)
              | ((tile_d0 & 0xffffu) << 16));                       // dim1 hi|tile0
  g1[4] = (int)(tile_d1 & 0xffffu);                                 // tile1 (tile2=0)
  g1[5] = (int)stride_elems;                                        // dim0_stride lo
  g1[6] = 0;
  g1[7] = 0;
#if HAVE_TDM
#if defined(__clang_major__) && __clang_major__ >= 23
  __builtin_amdgcn_tensor_load_to_lds(g0, g1, (v4i)0, (v4i)0, (v8i)0, 0);
#else
  __builtin_amdgcn_tensor_load_to_lds(g0, g1, (v4i)0, (v4i)0, 0);
#endif
#endif
}

// ---------------------------------------------------------------------------
// fp32 -> bf16 elementwise convert (n is a multiple of 1024)
// ---------------------------------------------------------------------------
__global__ void __launch_bounds__(256)
cvt_f32_to_bf16(const float* __restrict__ src, bf16* __restrict__ dst) {
  int i = (blockIdx.x * 256 + threadIdx.x) * 4;
  float4 v = *(const float4*)(src + i);
  dst[i + 0] = (bf16)v.x;
  dst[i + 1] = (bf16)v.y;
  dst[i + 2] = (bf16)v.z;
  dst[i + 3] = (bf16)v.w;
}

// pack bq,bk,bv -> one [768] fp32 bias vector (grid = 3, block = 256)
__global__ void __launch_bounds__(256)
pack_bias(const float* __restrict__ bq, const float* __restrict__ bk,
          const float* __restrict__ bv, float* __restrict__ dst) {
  const float* s = (blockIdx.x == 0) ? bq : (blockIdx.x == 1) ? bk : bv;
  dst[blockIdx.x * 256 + threadIdx.x] = s[threadIdx.x];
}

// ---------------------------------------------------------------------------
// Implicit-GEMM 3x3 SAME conv + bias + ReLU, bf16 operands, f32 WMMA accum.
// Double-buffered LDS; weight tile staged via async-to-LDS.
//   MODE 0: input x [B][C][N]; M=768 fused QKV. Q,K written TRANSPOSED
//           [B][N][C] (vectorized 8-channel stores); V written [B][C][N].
//   MODE 1: input ctx^T [B][N][C]; M=256 output conv -> fp32 [B][C][N] d_out.
// Block: 256 threads = 8 waves. Wave w -> 16(M) x 64(N) tile, 4 v8f accums.
// Grid : (64 image rows, M/128, B)
// ---------------------------------------------------------------------------
template <int MODE>
__global__ void __launch_bounds__(256, 1)
conv_gemm(const bf16* __restrict__ Wm, const bf16* __restrict__ Xb,
          const float* __restrict__ bias,
          bf16* __restrict__ outQ, bf16* __restrict__ outK,
          bf16* __restrict__ outV, float* __restrict__ outf) {
  constexpr int AR = 40;   // padded row stride (elems): 80B rows -> distinct banks
  __shared__ alignas(64) bf16 Atile[2][128][AR];  // [m][k]        20 KB
  __shared__ alignas(64) bf16 Bt[2][64][AR];      // [n][k] (K-T)  10 KB

  const int t    = threadIdx.x;
  const int lane = t & 31;
  const int wave = t >> 5;
  const int hi   = lane >> 4;    // lane-half selects K sub-block (ISA 7.12.2)
  const int ln   = lane & 15;

  const int b  = blockIdx.z;
  const int m0 = blockIdx.y * 128;
  const int h  = blockIdx.x;                    // 64-wide N tile == one image row
  const bf16* X = Xb + (size_t)b * C_ * N_;

  auto stageA = [&](int k0, int buf) {
    int r  = t >> 1;
    int kk = (t & 1) * 16;
    const bf16* src = Wm + (size_t)(m0 + r) * K9_ + k0 + kk;
    __builtin_prefetch(src + 64, 0, 1);          // global_prefetch_b8
    async_copy32(&Atile[buf][r][kk], src);
  };
  auto stageB = [&](int k0, int buf) {
    int n  = t >> 2;          // output x coordinate 0..63
    int j0 = (t & 3) * 8;     // k sub-offset
    v8bf tmp;
#pragma unroll
    for (int j = 0; j < 8; ++j) {
      int k   = k0 + j0 + j;
      int ic  = k / 9, rem = k % 9;
      int dy  = rem / 3 - 1, dx = rem % 3 - 1;
      int y   = h + dy, x = n + dx;
      bf16 v  = (bf16)0.0f;
      if ((unsigned)y < (unsigned)H_ && (unsigned)x < (unsigned)W_) {
        if (MODE == 0) v = X[(size_t)ic * N_ + y * W_ + x];       // [C][N]
        else           v = X[(size_t)(y * W_ + x) * C_ + ic];     // [N][C]
      }
      tmp[j] = v;
    }
    *(v8bf*)&Bt[buf][n][j0] = tmp;
  };

  v8f acc[4] = {};

  stageA(0, 0);
  stageB(0, 0);
  wait_async0();
  __syncthreads();

  for (int k0 = 0; k0 < K9_; k0 += 32) {
    const int cur = (k0 >> 5) & 1;
    if (k0 + 32 < K9_) {         // prefetch next tile while computing this one
      stageA(k0 + 32, cur ^ 1);
      stageB(k0 + 32, cur ^ 1);
    }
    // ---- A fragment: 16x32, lane m = ln, K halves split by lane-half -----
    v16bf a;
    {
      const bf16* Ar = &Atile[cur][wave * 16 + ln][0];
      v8bf lo = *(const v8bf*)&Ar[hi * 8];       // K  {0..7}  / {8..15}
      v8bf hb = *(const v8bf*)&Ar[16 + hi * 8];  // K {16..23} / {24..31}
#pragma unroll
      for (int j = 0; j < 8; ++j) { a[j] = lo[j]; a[j + 8] = hb[j]; }
    }
#pragma unroll
    for (int ch = 0; ch < 4; ++ch) {
      const bf16* Br = &Bt[cur][ch * 16 + ln][0];
      v8bf bl = *(const v8bf*)&Br[hi * 8], bh = *(const v8bf*)&Br[16 + hi * 8];
      v16bf bfrag;
#pragma unroll
      for (int j = 0; j < 8; ++j) { bfrag[j] = bl[j]; bfrag[j + 8] = bh[j]; }
      acc[ch] = wmma_bf16(a, bfrag, acc[ch]);
    }
    wait_async0();
    __syncthreads();
  }

  // ---- epilogue: bias + ReLU; C/D layout: M = r + 8*hi, N = ln -----------
  const int mb = m0 + wave * 16 + hi * 8;        // 8 consecutive output rows
#pragma unroll
  for (int ch = 0; ch < 4; ++ch) {
    int n  = ch * 16 + ln;
    int px = h * W_ + n;
    float vals[8];
#pragma unroll
    for (int r = 0; r < 8; ++r) {
      float v = acc[ch][r] + bias[mb + r];
      vals[r] = v > 0.0f ? v : 0.0f;
    }
    if (MODE == 0) {
      int proj = mb >> 8, oc0 = mb & 255;
      if (proj == 2) {
#pragma unroll
        for (int r = 0; r < 8; ++r)
          outV[((size_t)b * C_ + oc0 + r) * N_ + px] = (bf16)vals[r];
      } else {
        v8bf pack;
#pragma unroll
        for (int r = 0; r < 8; ++r) pack[r] = (bf16)vals[r];
        bf16* dst = (proj ? outK : outQ) + ((size_t)b * N_ + px) * C_ + oc0;
        *(v8bf*)dst = pack;                      // 16B store, 8 channels
      }
    } else {
#pragma unroll
      for (int r = 0; r < 8; ++r)
        outf[((size_t)b * C_ + mb + r) * N_ + px] = vals[r];
    }
  }
}

// ---------------------------------------------------------------------------
// Flash attention over channels: sim = softmax(Q^T K), ctx^T = 2 * (sim @ V^T).
//   qT,kT : [B][N][C] bf16 (channel-contiguous)  V : [B][C][N] bf16
//   ctxT  : [B][N][C] bf16 output (lane-contiguous channel stores)
// Block: 256 threads = 8 waves; wave owns 16 queries x full 256 channels.
// Grid : (N/128 = 32, B). Keys in blocks of 32, double-buffered TDM tiles.
// Dynamic LDS: Kt[2][32][264] + Vt[2][256][40] + Ps[8][16][32] = 82,944 B.
// ---------------------------------------------------------------------------
#define KTR 264   // Kt row stride (528B = 128+4 dwords): lanes distinct mod 256B
#define VTR 40    // Vt row stride  (80B =  16+4 dwords): lanes distinct mod 256B
#define FLASH_LDS ((2 * 32 * KTR + 2 * C_ * VTR + 8 * 16 * 32) * 2)

__global__ void __launch_bounds__(256, 1)
flash_attn(const bf16* __restrict__ qT, const bf16* __restrict__ kT,
           const bf16* __restrict__ V, bf16* __restrict__ ctxT) {
  extern __shared__ char smem[];
  bf16* KtB = (bf16*)smem;               // [2][32][KTR]  [key][c]
  bf16* VtB = KtB + 2 * 32 * KTR;        // [2][256][VTR] [c][key]
  bf16* PsB = VtB + 2 * C_ * VTR;        // [8][16][32]   per-wave P

  const int t    = threadIdx.x;
  const int lane = t & 31;
  const int wave = t >> 5;
  const int hi   = lane >> 4;
  const int ln   = lane & 15;

  const int b  = blockIdx.y;
  const int q0 = blockIdx.x * 128 + wave * 16;

  const bf16* kTb = kT + (size_t)b * N_ * C_;
  const bf16* Vb  = V + (size_t)b * C_ * N_;

  auto stage = [&](int kb, int buf) {
#if HAVE_TDM
    if (t == 0) {     // TDM: one wave moves both tiles; LDS pad via descriptor
      tdm_load_2d(KtB + (size_t)buf * 32 * KTR, kTb + (size_t)kb * C_,
                  /*tile*/ C_, 32, /*tensor*/ C_, N_, /*stride*/ C_,
                  /*pad: every 128 dwords (code 6) add 4 dwords (code 3)*/ 6, 3);
      tdm_load_2d(VtB + (size_t)buf * C_ * VTR, Vb + kb,
                  /*tile*/ 32, C_, /*tensor*/ N_, C_, /*stride*/ N_,
                  /*pad: every 16 dwords (code 3) add 4 dwords (code 3)*/ 3, 3);
    }
#else
    {  // async fallback: K tile rows are contiguous 512B, V rows 64B
      int key = t >> 3, ckk = (t & 7) * 32;
      async_copy64(KtB + (size_t)(buf * 32 + key) * KTR + ckk,
                   kTb + (size_t)(kb + key) * C_ + ckk);
      async_copy64(VtB + (size_t)(buf * C_ + t) * VTR, Vb + (size_t)t * N_ + kb);
    }
#endif
  };
  auto stage_wait = [&]() {
#if HAVE_TDM
    if (t == 0) __builtin_amdgcn_s_wait_tensorcnt(0);
#else
    wait_async0();
#endif
  };

  // ---- preload Q A-fragments from qT (contiguous 16B loads) ---------------
  const bf16* Qrow = qT + ((size_t)b * N_ + q0 + ln) * C_;
  v16bf qf[8];
#pragma unroll
  for (int kc = 0; kc < 8; ++kc) {
    v8bf lo = *(const v8bf*)&Qrow[kc * 32 + hi * 8];        // c {0..7}/{8..15}
    v8bf hb = *(const v8bf*)&Qrow[kc * 32 + 16 + hi * 8];   // c {16..23}/{24..31}
    v16bf a;
#pragma unroll
    for (int j = 0; j < 8; ++j) { a[j] = lo[j]; a[j + 8] = hb[j]; }
    qf[kc] = a;
  }

  v8f o[16];
#pragma unroll
  for (int ch = 0; ch < 16; ++ch) o[ch] = (v8f){};
  float mrun[8], lrun[8];
#pragma unroll
  for (int r = 0; r < 8; ++r) { mrun[r] = -3.0e38f; lrun[r] = 0.0f; }

  stage(0, 0);
  stage_wait();
  __syncthreads();

  for (int kb = 0; kb < N_; kb += 32) {
    const int cur = (kb >> 5) & 1;
    if (kb + 32 < N_) stage(kb + 32, cur ^ 1);   // overlap DMA with compute

    // ---- S = Q K^T : two 16x16 tiles (keys +0..15, +16..31) --------------
    v8f s0 = (v8f){}, s1 = (v8f){};
    const bf16* Kc = KtB + (size_t)cur * 32 * KTR;
#pragma unroll
    for (int kc = 0; kc < 8; ++kc) {
      const bf16* K0 = &Kc[(size_t)ln * KTR + kc * 32];
      const bf16* K1 = &Kc[(size_t)(16 + ln) * KTR + kc * 32];
      v8bf b0l = *(const v8bf*)&K0[hi * 8], b0h = *(const v8bf*)&K0[16 + hi * 8];
      v8bf b1l = *(const v8bf*)&K1[hi * 8], b1h = *(const v8bf*)&K1[16 + hi * 8];
      v16bf b0, b1;
#pragma unroll
      for (int j = 0; j < 8; ++j) {
        b0[j] = b0l[j]; b0[j + 8] = b0h[j]; b1[j] = b1l[j]; b1[j + 8] = b1h[j];
      }
      s0 = wmma_bf16(qf[kc], b0, s0);
      s1 = wmma_bf16(qf[kc], b1, s1);
    }

    // ---- online softmax (row = query = r + 8*hi, 16 lanes per row) -------
    bf16* Pw = PsB + (size_t)wave * 16 * 32;
    float scl[8];
#pragma unroll
    for (int r = 0; r < 8; ++r) {
      float mx = fmaxf(s0[r], s1[r]);
#pragma unroll
      for (int d = 1; d < 16; d <<= 1) mx = fmaxf(mx, __shfl_xor(mx, d, 32));
      float nm = fmaxf(mrun[r], mx);
      float sc = __expf(mrun[r] - nm);
      float p0 = __expf(s0[r] - nm);
      float p1 = __expf(s1[r] - nm);
      int m = r + hi * 8;
      Pw[m * 32 + ln]      = (bf16)p0;   // same-wave LDS: DS ops are in-order,
      Pw[m * 32 + 16 + ln] = (bf16)p1;   // no barrier needed before reload
      float rs = p0 + p1;
#pragma unroll
      for (int d = 1; d < 16; d <<= 1) rs += __shfl_xor(rs, d, 32);
      lrun[r] = lrun[r] * sc + rs;
      mrun[r] = nm;
      scl[r]  = sc;
    }
#pragma unroll
    for (int ch = 0; ch < 16; ++ch)
#pragma unroll
      for (int r = 0; r < 8; ++r) o[ch][r] *= scl[r];

    // ---- P A-fragment (reused across all 16 channel chunks) --------------
    v16bf pa;
    {
      const bf16* Pr = Pw + ln * 32;
      v8bf lo = *(const v8bf*)&Pr[hi * 8];
      v8bf hb = *(const v8bf*)&Pr[16 + hi * 8];
#pragma unroll
      for (int j = 0; j < 8; ++j) { pa[j] = lo[j]; pa[j + 8] = hb[j]; }
    }
    // ---- O(16x256) += P(16x32) @ V(32x256) : 16 WMMAs --------------------
    const bf16* Vc = VtB + (size_t)cur * C_ * VTR;
#pragma unroll
    for (int ch = 0; ch < 16; ++ch) {
      const bf16* Vr = &Vc[(size_t)(ch * 16 + ln) * VTR];
      v8bf vl = *(const v8bf*)&Vr[hi * 8], vh = *(const v8bf*)&Vr[16 + hi * 8];
      v16bf bv;
#pragma unroll
      for (int j = 0; j < 8; ++j) { bv[j] = vl[j]; bv[j + 8] = vh[j]; }
      o[ch] = wmma_bf16(pa, bv, o[ch]);
    }

    stage_wait();       // next tile landed, then make it block-visible
    __syncthreads();
  }

  // ---- epilogue: ctx^T = 2 * O / l ; lanes write 16 consecutive channels --
#pragma unroll
  for (int ch = 0; ch < 16; ++ch)
#pragma unroll
    for (int r = 0; r < 8; ++r) {
      int m = r + hi * 8;                // query row within wave tile
      int c = ch * 16 + ln;              // channel
      float v = o[ch][r] * (2.0f / lrun[r]);
      ctxT[((size_t)b * N_ + q0 + m) * C_ + c] = (bf16)v;
    }
}

// ---------------------------------------------------------------------------
// Launch pipeline. Workspace use: ~47 MB.
// ---------------------------------------------------------------------------
extern "C" void kernel_launch(void* const* d_in, const int* in_sizes, int n_in,
                              void* d_out, int out_size, void* d_ws, size_t ws_size,
                              hipStream_t stream) {
  const float* x  = (const float*)d_in[0];
  const float* wq = (const float*)d_in[1];
  const float* bq = (const float*)d_in[2];
  const float* wk = (const float*)d_in[3];
  const float* bk = (const float*)d_in[4];
  const float* wv = (const float*)d_in[5];
  const float* bv = (const float*)d_in[6];
  const float* wo = (const float*)d_in[7];
  const float* bo = (const float*)d_in[8];
  float* out = (float*)d_out;

  const size_t nX = (size_t)B_ * C_ * N_;   // 4194304
  const size_t nW = (size_t)C_ * K9_;       // 589824

  char* ws = (char*)d_ws;
  size_t off = 0;
  auto carve = [&](size_t bytes) -> char* {
    char* p = ws + off;
    off = (off + bytes + 255) & ~(size_t)255;
    return p;
  };
  bf16*  xb    = (bf16*)carve(nX * sizeof(bf16));     // x in bf16 [B][C][N]
  bf16*  qTd   = (bf16*)carve(nX * sizeof(bf16));     // Q^T  [B][N][C]
  bf16*  kTd   = (bf16*)carve(nX * sizeof(bf16));     // K^T  [B][N][C]
  bf16*  vd    = (bf16*)carve(nX * sizeof(bf16));     // V    [B][C][N]
  bf16*  ctxT  = (bf16*)carve(nX * sizeof(bf16));     // ctx^T[B][N][C]
  bf16*  wqkvb = (bf16*)carve(3 * nW * sizeof(bf16)); // [768][2304]
  bf16*  wob   = (bf16*)carve(nW * sizeof(bf16));     // [256][2304]
  float* bqkv  = (float*)carve(768 * sizeof(float));  // packed biases
  (void)ws_size; (void)in_sizes; (void)n_in; (void)out_size;

  // 1) precision conversion / packing
  cvt_f32_to_bf16<<<dim3(nX / 1024), dim3(256), 0, stream>>>(x, xb);
  cvt_f32_to_bf16<<<dim3(nW / 1024), dim3(256), 0, stream>>>(wq, wqkvb);
  cvt_f32_to_bf16<<<dim3(nW / 1024), dim3(256), 0, stream>>>(wk, wqkvb + nW);
  cvt_f32_to_bf16<<<dim3(nW / 1024), dim3(256), 0, stream>>>(wv, wqkvb + 2 * nW);
  cvt_f32_to_bf16<<<dim3(nW / 1024), dim3(256), 0, stream>>>(wo, wob);
  pack_bias<<<dim3(3), dim3(256), 0, stream>>>(bq, bk, bv, bqkv);

  // 2) fused QKV 3x3 conv (implicit GEMM, M=768) + bias + ReLU -> bf16
  //    Q,K transposed for contiguous TDM/async flash staging; V channel-major.
  conv_gemm<0><<<dim3(H_, 768 / 128, B_), dim3(256), 0, stream>>>(
      wqkvb, xb, bqkv, qTd, kTd, vd, nullptr);

  // 3) flash attention: softmax(Q^T K) @ V^T, x2 fold -> bf16 ctx^T
  flash_attn<<<dim3(N_ / 128, B_), dim3(256), FLASH_LDS, stream>>>(
      qTd, kTd, vd, ctxT);

  // 4) output 3x3 conv (M=256, reads ctx^T) + bias + ReLU -> fp32 d_out
  conv_gemm<1><<<dim3(H_, 256 / 128, B_), dim3(256), 0, stream>>>(
      wob, ctxT, bo, nullptr, nullptr, nullptr, out);
}
